// MaskedLstm_15676630630966
// MI455X (gfx1250) — matrix operand order repaired and checked by reading, not compiled, not run on hardware
//
#include <hip/hip_runtime.h>

// ---------------------------------------------------------------------------
// Masked LSTM (L0 hard-concrete gates), T=512 B=64 I=512 H=768, seq-first.
// Persistent-RNN design for MI455X (gfx1250, wave32, WMMA):
//   - prep kernels: fp32 -> bf16 casts of x/W/U, precompute s[t][h]
//   - one persistent scan kernel: 48 WGs x 256 threads, each WG owns 16
//     hidden units (64 gate columns). W/U slices pinned in LDS (160KB of the
//     320KB WGP LDS). Per step: fused GEMM [64,1280]x[1280,64] via
//     v_wmma_f32_16x16x32_bf16, epilogue through LDS, cell state in regs.
//   - SPLIT grid barrier: arrive after publishing h*s, wait only before the
//     h-dependent GEMM phase, so the x@W phase of step t+1 overlaps the
//     barrier latency of step t (the scan is critical-path bound, not BW).
// ---------------------------------------------------------------------------

#define T_STEPS 512
#define BATCH   64
#define IN_DIM  512
#define HID     768
#define G4      (4 * HID)   // 3072

#define NWG     48          // workgroups (each owns HID/NWG = 16 hidden units)
#define HS      16          // hidden units per WG
#define NCOLS   64          // gate columns per WG (4 gates x 16)
#define THREADS 256         // 8 waves

typedef __attribute__((ext_vector_type(16))) __bf16         v16bf;
typedef __attribute__((ext_vector_type(8)))  float          v8f;
typedef __attribute__((ext_vector_type(4)))  float          v4f;
typedef __attribute__((ext_vector_type(4)))  unsigned int   v4u;
typedef __attribute__((ext_vector_type(4)))  unsigned short v4us;

union FragBf {      // 32 bytes: one WMMA bf16 A/B fragment per lane
    v4u   q[2];
    v16bf v;
};

__device__ __forceinline__ unsigned short f2bf(float f) {
    union { float f; unsigned u; } cv; cv.f = f;
    unsigned u = cv.u;
    unsigned r = u + 0x7FFFu + ((u >> 16) & 1u);   // round-to-nearest-even
    return (unsigned short)(r >> 16);
}

__device__ __forceinline__ float sigmoidf_(float x) {
    return 1.0f / (1.0f + __expf(-x));
}

// ------------------------------ prep kernels -------------------------------

__global__ void cast_f32_to_bf16(const float* __restrict__ src,
                                 unsigned short* __restrict__ dst, int n) {
    for (int i = blockIdx.x * blockDim.x + threadIdx.x; i < n;
         i += gridDim.x * blockDim.x)
        dst[i] = f2bf(src[i]);
}

__global__ void gate_s_kernel(const float* __restrict__ u,
                              const float* __restrict__ log_alpha,
                              float* __restrict__ s, int n) {
    const float INV_BETA = 1.5f;        // 1/(2/3)
    const float GAMMA = -0.1f, ZETA = 1.1f;
    for (int i = blockIdx.x * blockDim.x + threadIdx.x; i < n;
         i += gridDim.x * blockDim.x) {
        int j = i % HID;
        float uv  = u[i];
        float lg  = (__logf(uv) - log1pf(-uv) + log_alpha[j]) * INV_BETA;
        float sig = 1.0f / (1.0f + __expf(-lg));
        float sv  = sig * (ZETA - GAMMA) + GAMMA;
        s[i] = fminf(1.0f, fmaxf(0.0f, sv));
    }
}

// ------------------------------ scan kernel --------------------------------

__global__ void __launch_bounds__(THREADS, 1)
lstm_scan(const unsigned short* __restrict__ xbf,   // [T*B, I] bf16
          const unsigned short* __restrict__ wbf,   // [4H, I]  bf16
          const unsigned short* __restrict__ ubf,   // [4H, H]  bf16
          const float* __restrict__ bias,           // [4H]
          const float* __restrict__ s_all,          // [T, H]
          unsigned short* __restrict__ hbuf0,       // [B, H] bf16 (h*s) ping
          unsigned short* __restrict__ hbuf1,       // [B, H] bf16 (h*s) pong
          float* __restrict__ out,                  // [T, B, H]
          unsigned* __restrict__ bar_cnt,
          unsigned* __restrict__ bar_gen)
{
    extern __shared__ char smem[];
    unsigned short* sW = (unsigned short*)smem;                        // 64*512 bf16 = 64KB
    unsigned short* sU = (unsigned short*)(smem + 64 * IN_DIM * 2);    // 64*768 bf16 = 96KB
    float*          sG = (float*)(smem + 64 * IN_DIM * 2 + 64 * HID * 2); // 64*64 f32 = 16KB

    const int tid   = threadIdx.x;
    const int hbase = blockIdx.x * HS;
    const int lane  = tid & 31;
    const int wid   = tid >> 5;
    const int nlane = lane & 15;
    const int lhalf = lane >> 4;

    // ---- stage this WG's weight slices into LDS (once, reused 512 steps) ----
    for (int idx = tid; idx < 64 * (IN_DIM / 8); idx += THREADS) {
        int row = idx >> 6, chunk = idx & 63;               // IN_DIM/8 = 64 chunks
        int q = row >> 4, jj = row & 15;
        const v4u* src = (const v4u*)(wbf + (size_t)(q * HID + hbase + jj) * IN_DIM) + chunk;
        ((v4u*)sW)[row * 64 + chunk] = *src;
    }
    for (int idx = tid; idx < 64 * (HID / 8); idx += THREADS) {
        int row = idx / 96, chunk = idx - row * 96;         // HID/8 = 96 chunks
        int q = row >> 4, jj = row & 15;
        const v4u* src = (const v4u*)(ubf + (size_t)(q * HID + hbase + jj) * HID) + chunk;
        ((v4u*)sU)[row * 96 + chunk] = *src;
    }
    __syncthreads();

    // wave -> output-tile assignment: 4 M-tiles (batch) x 4 N-tiles (gates)
    const int mtile = wid & 3;
    const int nt0   = (wid >> 2) * 2;
    const int nt1   = nt0 + 1;

    const float bv0 = bias[nt0 * HID + hbase + nlane];
    const float bv1 = bias[nt1 * HID + hbase + nlane];

    // per-thread cell state: thread owns 4 consecutive (batch,hidden) elems
    const int e0  = tid * 4;
    const int ubb = e0 >> 4;     // batch row
    const int ujj = e0 & 15;     // first local hidden index
    float creg[4] = {0.f, 0.f, 0.f, 0.f};

    const int arow   = mtile * 16 + nlane;  // A-matrix row = batch index
    const int khalf8 = lhalf * 16;          // A fragment lane-half K offset

    for (int t = 0; t < T_STEPS; ++t) {
        const unsigned short* ping = (t & 1) ? hbuf1 : hbuf0;
        unsigned short*       pong = (t & 1) ? hbuf0 : hbuf1;

        v8f acc0 = {};
        v8f acc1 = {};

        // ---- phase 1: x_t @ W^T  (K = 512) — independent of h, runs while
        //      other workgroups may still be finishing step t-1 ----
        const unsigned short* ax = xbf + ((size_t)t * BATCH + arow) * IN_DIM;
        #pragma unroll 4
        for (int kb = 0; kb < IN_DIM; kb += 32) {
            FragBf a, f0, f1;
            a.q[0] = *(const v4u*)(ax + kb + khalf8);
            a.q[1] = *(const v4u*)(ax + kb + khalf8 + 8);
            const unsigned short* p0 = sW + (nt0 * 16 + nlane) * IN_DIM + kb + lhalf * 16;
            const unsigned short* p1 = sW + (nt1 * 16 + nlane) * IN_DIM + kb + lhalf * 16;
            f0.q[0] = *(const v4u*)(p0);     f0.q[1] = *(const v4u*)(p0 + 8);
            f1.q[0] = *(const v4u*)(p1);     f1.q[1] = *(const v4u*)(p1 + 8);
            acc0 = __builtin_amdgcn_wmma_f32_16x16x32_bf16(false, a.v, false, f0.v,
                                                           (short)0, acc0, false, false);
            acc1 = __builtin_amdgcn_wmma_f32_16x16x32_bf16(false, a.v, false, f1.v,
                                                           (short)0, acc1, false, false);
        }

        // ---- barrier WAIT: h_{t-1}*s_t published by all WGs (gen >= t) ----
        if (tid == 0) {
            while (__hip_atomic_load(bar_gen, __ATOMIC_ACQUIRE,
                                     __HIP_MEMORY_SCOPE_AGENT) < (unsigned)t) {
                __builtin_amdgcn_s_sleep(2);
            }
        }
        __syncthreads();

        // ---- phase 2: (h_{t-1} * s_t) @ U^T  (K = 768) ----
        const unsigned short* ah = ping + (size_t)arow * HID;
        #pragma unroll 4
        for (int kb = 0; kb < HID; kb += 32) {
            FragBf a, f0, f1;
            a.q[0] = *(const v4u*)(ah + kb + khalf8);
            a.q[1] = *(const v4u*)(ah + kb + khalf8 + 8);
            const unsigned short* p0 = sU + (nt0 * 16 + nlane) * HID + kb + lhalf * 16;
            const unsigned short* p1 = sU + (nt1 * 16 + nlane) * HID + kb + lhalf * 16;
            f0.q[0] = *(const v4u*)(p0);     f0.q[1] = *(const v4u*)(p0 + 8);
            f1.q[0] = *(const v4u*)(p1);     f1.q[1] = *(const v4u*)(p1 + 8);
            acc0 = __builtin_amdgcn_wmma_f32_16x16x32_bf16(false, a.v, false, f0.v,
                                                           (short)0, acc0, false, false);
            acc1 = __builtin_amdgcn_wmma_f32_16x16x32_bf16(false, a.v, false, f1.v,
                                                           (short)0, acc1, false, false);
        }

        // ---- epilogue: gates_pre = (acc + b) * s_t  -> LDS ----
        // C/D layout: lane n = lane%16, VGPR r -> row r + 8*(lane/16)
        const float sv = s_all[(size_t)t * HID + hbase + nlane];
        #pragma unroll
        for (int r = 0; r < 8; ++r) {
            int bb = mtile * 16 + lhalf * 8 + r;
            sG[bb * NCOLS + nt0 * 16 + nlane] = (acc0[r] + bv0) * sv;
            sG[bb * NCOLS + nt1 * 16 + nlane] = (acc1[r] + bv1) * sv;
        }
        __syncthreads();

        // ---- elementwise state update (c in registers) ----
        v4f  hv;
        v4us pv;
        #pragma unroll
        for (int i = 0; i < 4; ++i) {
            int j = ujj + i;
            float svj = s_all[(size_t)t * HID + hbase + j];
            float gi = sG[ubb * NCOLS +      j];
            float gf = sG[ubb * NCOLS + 16 + j];
            float gc = sG[ubb * NCOLS + 32 + j];
            float go = sG[ubb * NCOLS + 48 + j];
            float ig = sigmoidf_(gi) * svj;
            float fg = sigmoidf_(gf) * svj;
            float cg = tanhf(gc)     * svj;
            float og = sigmoidf_(go) * svj;
            creg[i] = fg * creg[i] + ig * cg;
            float h = og * tanhf(creg[i]);
            hv[i] = h;
            if (t + 1 < T_STEPS) {
                float sn = s_all[(size_t)(t + 1) * HID + hbase + j];
                pv[i] = f2bf(h * sn);
            }
        }
        *(v4f*)(out + (size_t)t * (BATCH * HID) + (size_t)ubb * HID + hbase + ujj) = hv;
        if (t + 1 < T_STEPS) {
            *(v4us*)(pong + (size_t)ubb * HID + hbase + ujj) = pv;
        }

        // ---- barrier ARRIVE: publish h_t * s_{t+1}; do NOT wait here ----
        __syncthreads();   // all pong/out writes issued; LDS reads done
        if (tid == 0) {
            __threadfence();
            unsigned prev = __hip_atomic_fetch_add(bar_cnt, 1u, __ATOMIC_ACQ_REL,
                                                   __HIP_MEMORY_SCOPE_AGENT);
            if (prev == (unsigned)(NWG * (t + 1) - 1)) {
                __hip_atomic_store(bar_gen, (unsigned)(t + 1), __ATOMIC_RELEASE,
                                   __HIP_MEMORY_SCOPE_AGENT);
            }
        }
        // next iteration's phase 1 (x@W) proceeds immediately
    }
}

// ------------------------------ host launcher ------------------------------

extern "C" void kernel_launch(void* const* d_in, const int* in_sizes, int n_in,
                              void* d_out, int out_size, void* d_ws, size_t ws_size,
                              hipStream_t stream) {
    (void)in_sizes; (void)n_in; (void)out_size; (void)ws_size;

    const float* x  = (const float*)d_in[0];   // [T,B,I]
    const float* W  = (const float*)d_in[1];   // [4H,I]
    const float* U  = (const float*)d_in[2];   // [4H,H]
    const float* b  = (const float*)d_in[3];   // [4H]
    const float* la = (const float*)d_in[4];   // [H]
    const float* u  = (const float*)d_in[5];   // [T,H]

    // workspace layout (all 16B-aligned); total ~43.2 MB
    char* ws = (char*)d_ws;
    unsigned*       bar   = (unsigned*)ws;                                  // 256 B
    unsigned short* hbuf0 = (unsigned short*)(ws + 256);                    // 96 KB
    unsigned short* hbuf1 = (unsigned short*)(ws + 256 + BATCH * HID * 2);  // 96 KB
    unsigned short* wbf   = (unsigned short*)(ws + 256 + 2 * BATCH * HID * 2);
    unsigned short* ubf   = wbf + (size_t)G4 * IN_DIM;
    unsigned short* xbf   = ubf + (size_t)G4 * HID;
    float*          s_all = (float*)(xbf + (size_t)T_STEPS * BATCH * IN_DIM);

    // zero barrier state + h*s ping buffer (h0 == 0) — every call
    hipMemsetAsync(ws, 0, 256 + BATCH * HID * 2, stream);

    cast_f32_to_bf16<<<2048, 256, 0, stream>>>(W, wbf, G4 * IN_DIM);
    cast_f32_to_bf16<<<2048, 256, 0, stream>>>(U, ubf, G4 * HID);
    cast_f32_to_bf16<<<4096, 256, 0, stream>>>(x, xbf, T_STEPS * BATCH * IN_DIM);
    gate_s_kernel  <<<1536, 256, 0, stream>>>(u, la, s_all, T_STEPS * HID);

    size_t smem = (size_t)64 * IN_DIM * 2 + (size_t)64 * HID * 2 +
                  (size_t)64 * 64 * 4;   // 64KB W + 96KB U + 16KB gates = 176KB
    lstm_scan<<<NWG, THREADS, smem, stream>>>(xbf, wbf, ubf, b, s_all,
                                              hbuf0, hbuf1, (float*)d_out,
                                              bar, bar + 1);
}
